// PixelCNNConditionalDistribution_37735582663394
// MI455X (gfx1250) — compile-verified, matrix operand order checked by reading, not run on hardware
//
#include <hip/hip_runtime.h>
#include <math.h>

typedef __attribute__((ext_vector_type(16))) _Float16 v16h;
typedef __attribute__((ext_vector_type(8)))  float    v8f;

#ifndef TRY_TDM
#define TRY_TDM 1
#endif
#if TRY_TDM && defined(__has_builtin)
#  if __has_builtin(__builtin_amdgcn_tensor_load_to_lds)
#    define USE_TDM 1
#  else
#    define USE_TDM 0
#  endif
#else
#  define USE_TDM 0
#endif

typedef unsigned int v4u  __attribute__((ext_vector_type(4)));
typedef int          v8i_ __attribute__((ext_vector_type(8)));
typedef int          v4i_ __attribute__((ext_vector_type(4)));

// LDS layout (dynamic): As = 128 x 64 f16 (16 KB) @ 0, Bs = 64 x 32 f16 (4 KB) @ 16384
#define LDS_BYTES 20480
#define BS_OFF    16384

// ---------------------------------------------------------------------------
// Implicit-GEMM convolution, WMMA f16->f32 core, fused activation.
//   Y[M, Cout] (+)= act(X) (*) Wh + bias    (Wh = f16 weights, row-major [K,Cout])
//   act: 0 = none, 1 = concat_elu (Cin = 2*Cx), 2 = elu
// Block: 256 threads = 8 waves; tile 128(M) x 32(N); K-step 64.
// Each wave: one 16-row A fragment, two N-subtiles -> 2 accumulators,
// 4 WMMA per K-step. B tiles staged via Tensor Data Mover when available.
// ---------------------------------------------------------------------------
__global__ __launch_bounds__(256)
void conv_wmma_k(const float* __restrict__ X, const _Float16* __restrict__ Wh,
                 const float* __restrict__ bias, float* __restrict__ Y,
                 int Bn, int H, int W, int Cx, int Cin,
                 int kh, int kw, int pad_t, int pad_l, int stride,
                 int Ho, int Wo, int Cout, int act, int accum)
{
    extern __shared__ char smem[];
    _Float16* As = (_Float16*)smem;             // [row][64]
    _Float16* Bs = (_Float16*)(smem + BS_OFF);  // [k][32]

    const int K  = kh * kw * Cin;
    const int M  = Bn * Ho * Wo;
    const int m0 = blockIdx.x * 128;
    const int n0 = blockIdx.y * 32;
    const int tid  = threadIdx.x;
    const int lane = tid & 31;
    const int wave = tid >> 5;

    // ---- per-thread fixed output row (A loader): row = tid>>1, khalf = (tid&1)*32
    const int arow  = tid >> 1;
    const int khalf = (tid & 1) << 5;
    const int m = m0 + arow;
    const bool mok = (m < M);
    int bN = 0, hi0 = 0, wi0 = 0;
    if (mok) {
        int hw  = Ho * Wo;
        bN      = m / hw;
        int rem = m - bN * hw;
        int ho  = rem / Wo;
        int wo  = rem - ho * Wo;
        hi0 = ho * stride - pad_t;
        wi0 = wo * stride - pad_l;
    }

    v8f acc0 = {}, acc1 = {};
    const int ksteps = (K + 63) >> 6;

    for (int ks = 0; ks < ksteps; ++ks) {
        const int kbase = ks << 6;

        // prefetch next K-step's weights (speculative; dropped if invalid)
        __builtin_prefetch(Wh + (size_t)(kbase + 64) * Cout + n0 + lane, 0, 1);

        // ---- stage A half-row (32 elems), incremental (ki,kj,c) walk -------
        {
            int k0 = kbase + khalf;
            int c    = k0 % Cin;
            int rest = k0 / Cin;
            int kj   = rest % kw;
            int ki   = rest / kw;
            _Float16* adst = As + arow * 64 + khalf;
            #pragma unroll 8
            for (int t = 0; t < 32; ++t) {
                float v = 0.f;
                if (mok && (k0 + t) < K) {
                    int hi = hi0 + ki, wi = wi0 + kj;
                    if (hi >= 0 && hi < H && wi >= 0 && wi < W) {
                        int cc = (act == 1 && c >= Cx) ? (c - Cx) : c;
                        float xv = X[((size_t)(bN * H + hi) * W + wi) * Cx + cc];
                        if (act == 1 && c >= Cx) xv = -xv;
                        if (act != 0) xv = xv > 0.f ? xv : __expf(xv) - 1.f;
                        v = xv;
                    }
                }
                adst[t] = (_Float16)v;
                if (++c == Cin) { c = 0; if (++kj == kw) { kj = 0; ++ki; } }
            }
        }

        // ---- stage B tile 64(k) x 32(n) -> Bs[k][n] ------------------------
#if USE_TDM
        if (tid == 0) {
            unsigned long long ga =
                (unsigned long long)(const void*)(Wh + (size_t)kbase * Cout + n0);
            unsigned rem_n = (unsigned)(Cout - n0);      // OOB zero-fill in N
            unsigned rem_k = (unsigned)(K - kbase);      // OOB zero-fill in K
            v4u g0;
            g0[0] = 1u;                                   // count=1, user mode
            g0[1] = (unsigned)BS_OFF;                     // lds_addr
            g0[2] = (unsigned)(ga & 0xffffffffull);       // global_addr[31:0]
            g0[3] = (unsigned)((ga >> 32) & 0x1ffffffu) | (2u << 30); // addr hi | type=2
            v8i_ g1;
            g1[0] = (int)(1u << 16);                      // workgroup_mask=0, data_size=2B
            g1[1] = (int)((rem_n & 0xffffu) << 16);       // tensor_dim0[15:0]
            g1[2] = (int)((rem_n >> 16) | ((rem_k & 0xffffu) << 16)); // dim0 hi | dim1 lo
            g1[3] = (int)((rem_k >> 16) | (32u << 16));   // dim1 hi | tile_dim0=32
            g1[4] = (int)64u;                             // tile_dim1=64, tile_dim2=0
            g1[5] = (int)(unsigned)Cout;                  // tensor_dim0_stride[31:0]
            g1[6] = 0;                                    // stride hi, dim1_stride lo
            g1[7] = 0;
            v4i_ z4 = {};
#  if __has_include(<hip/amd_detail/amd_gfx1250_TDM.h>)
            v8i_ z8 = {};
            __builtin_amdgcn_tensor_load_to_lds(g0, g1, z4, z4, z8, 0);
#  else
            __builtin_amdgcn_tensor_load_to_lds(g0, g1, z4, z4, 0);
#  endif
        }
        __builtin_amdgcn_s_wait_tensorcnt(0);
#else
        #pragma unroll
        for (int t = 0; t < 8; ++t) {
            int e  = t * 256 + tid;
            int kk = e >> 5, nn = e & 31;
            int k = kbase + kk, n = n0 + nn;
            Bs[kk * 32 + nn] = (k < K && n < Cout) ? Wh[(size_t)k * Cout + n]
                                                   : (_Float16)0.f;
        }
#endif
        __syncthreads();

        // ---- fragments + 4 WMMA (2 K-substeps x 2 N-subtiles) --------------
        const int rrow = (wave << 4) + (lane & 15);
        const int g    = lane >> 4;
        const int nsel = lane & 15;
        #pragma unroll
        for (int s = 0; s < 2; ++s) {
            v16h af = {}, bf0 = {}, bf1 = {};
            #pragma unroll
            for (int vr = 0; vr < 8; ++vr) {
                int kbl = (s << 5) + ((vr < 4) ? (vr * 2) : (16 + (vr - 4) * 2)) + (g << 3);
                af[2 * vr]      = As[rrow * 64 + kbl];
                af[2 * vr + 1]  = As[rrow * 64 + kbl + 1];
                bf0[2 * vr]     = Bs[kbl * 32 + nsel];
                bf0[2 * vr + 1] = Bs[(kbl + 1) * 32 + nsel];
                bf1[2 * vr]     = Bs[kbl * 32 + 16 + nsel];
                bf1[2 * vr + 1] = Bs[(kbl + 1) * 32 + 16 + nsel];
            }
            acc0 = __builtin_amdgcn_wmma_f32_16x16x32_f16(
                       false, af, false, bf0, (short)0, acc0, false, false);
            acc1 = __builtin_amdgcn_wmma_f32_16x16x32_f16(
                       false, af, false, bf1, (short)0, acc1, false, false);
        }
        __syncthreads();
    }

    // ---- epilogue: C/D layout lane->n=lane&15, row = vr + 8*(lane>>4) ------
    const int nl = lane & 15;
    const int g  = lane >> 4;
    #pragma unroll
    for (int vr = 0; vr < 8; ++vr) {
        int mo = m0 + (wave << 4) + vr + (g << 3);
        if (mo < M) {
            int n1 = n0 + nl;
            if (n1 < Cout) {
                float o = acc0[vr] + bias[n1];
                if (accum) o += Y[(size_t)mo * Cout + n1];
                Y[(size_t)mo * Cout + n1] = o;
            }
            int n2 = n0 + 16 + nl;
            if (n2 < Cout) {
                float o = acc1[vr] + bias[n2];
                if (accum) o += Y[(size_t)mo * Cout + n2];
                Y[(size_t)mo * Cout + n2] = o;
            }
        }
    }
}

// ---------------------------------------------------------------------------
// Small pointwise / helper kernels
// ---------------------------------------------------------------------------
__global__ void w2h_k(const float* __restrict__ w, _Float16* __restrict__ o, int n)
{
    int i = blockIdx.x * blockDim.x + threadIdx.x;
    if (i < n) o[i] = (_Float16)w[i];
}

__global__ void prep_k(const float* __restrict__ x, float* __restrict__ xp, int n)
{   // x: (4,3,32,32) NCHW -> xp: (4,32,32,4) NHWC, *2-1, last channel = 1
    int i = blockIdx.x * blockDim.x + threadIdx.x;
    if (i >= n) return;
    int c = i & 3, t = i >> 2;
    int w = t & 31; t >>= 5;
    int h = t & 31; int b = t >> 5;
    xp[i] = (c < 3) ? x[(((size_t)b * 3 + c) * 32 + h) * 32 + w] * 2.f - 1.f : 1.f;
}

__global__ void dshift_k(const float* __restrict__ in, float* __restrict__ o,
                         int Bn, int H, int W, int C, int n)
{
    int i = blockIdx.x * blockDim.x + threadIdx.x;
    if (i >= n) return;
    int c = i % C; int t = i / C;
    int w = t % W; t /= W;
    int h = t % H; int b = t / H;
    o[i] = (h > 0) ? in[(((size_t)b * H + (h - 1)) * W + w) * C + c] : 0.f;
}

__global__ void rshift_k(const float* __restrict__ in, float* __restrict__ o,
                         int Bn, int H, int W, int C, int n)
{
    int i = blockIdx.x * blockDim.x + threadIdx.x;
    if (i >= n) return;
    int c = i % C; int t = i / C;
    int w = t % W; t /= W;
    int h = t % H; int b = t / H;
    o[i] = (w > 0) ? in[(((size_t)b * H + h) * W + (w - 1)) * C + c] : 0.f;
}

__global__ void add_k(const float* a, const float* b, float* o, int n)
{
    int i = blockIdx.x * blockDim.x + threadIdx.x;
    if (i < n) o[i] = a[i] + b[i];
}

__global__ void concat_k(const float* __restrict__ a, const float* __restrict__ b,
                         float* __restrict__ o, int M, int C)
{
    int i = blockIdx.x * blockDim.x + threadIdx.x;
    if (i >= M * 2 * C) return;
    int m = i / (2 * C), ch = i % (2 * C);
    o[i] = (ch < C) ? a[(size_t)m * C + ch] : b[(size_t)m * C + (ch - C)];
}

__global__ void cond_mm_k(const float* __restrict__ h, const float* __restrict__ w,
                          const float* __restrict__ b, float* __restrict__ o,
                          int Bn, int Kc, int N)
{
    int i = blockIdx.x * blockDim.x + threadIdx.x;
    if (i >= Bn * N) return;
    int bb = i / N, n = i % N;
    float s = b[n];
    for (int k = 0; k < Kc; ++k) s += h[bb * Kc + k] * w[(size_t)k * N + n];
    o[i] = s;
}

__global__ void gate_k(const float* __restrict__ x, const float* __restrict__ c,
                       const float* __restrict__ cond, float* __restrict__ o,
                       int M, int C, int HW)
{
    int i = blockIdx.x * blockDim.x + threadIdx.x;
    if (i >= M * C) return;
    int m = i / C, ch = i % C;
    int b = m / HW;
    float g = c[(size_t)m * 2 * C + ch]      + cond[b * 2 * C + ch];
    float s = c[(size_t)m * 2 * C + C + ch]  + cond[b * 2 * C + C + ch];
    o[i] = x[i] + g * (1.f / (1.f + __expf(-s)));
}

// Transposed conv (lhs_dilation=2) with reference pad+crop folded in:
// out[h2,w2,co] = sum_{ki,kj,ci} w[ki,kj,ci,co] * x[(h2+ki-1)/2,(w2+kj-1)/2,ci]
__global__ void deconv_k(const float* __restrict__ X, const float* __restrict__ Wt,
                         const float* __restrict__ bias, float* __restrict__ Y,
                         int Bn, int H, int W, int C, int kh, int kw, int n)
{
    int i = blockIdx.x * blockDim.x + threadIdx.x;
    if (i >= n) return;
    int co = i % C; int t = i / C;
    int wo = t % (2 * W); t /= (2 * W);
    int ho = t % (2 * H); int b = t / (2 * H);
    float s = bias[co];
    for (int ki = 0; ki < kh; ++ki) {
        int rr = ho + ki - 1;
        if (rr < 0 || (rr & 1) || (rr >> 1) >= H) continue;
        int hi = rr >> 1;
        for (int kj = 0; kj < kw; ++kj) {
            int cc = wo + kj - 1;
            if (cc < 0 || (cc & 1) || (cc >> 1) >= W) continue;
            int wi = cc >> 1;
            const float* xr = X + (((size_t)b * H + hi) * W + wi) * C;
            const float* wr = Wt + ((size_t)(ki * kw + kj) * C) * C + co;
            for (int ci = 0; ci < C; ++ci) s += xr[ci] * wr[(size_t)ci * C];
        }
    }
    Y[i] = s;
}

__global__ void zero_k(float* p, int n)
{
    int i = blockIdx.x * blockDim.x + threadIdx.x;
    if (i < n) p[i] = 0.f;
}

// ---------------------------------------------------------------------------
// DMLL log-prob: one thread per pixel, 100 mixtures, atomicAdd per batch.
// ---------------------------------------------------------------------------
__device__ __forceinline__ float sp_(float z)
{   return fmaxf(z, 0.f) + log1pf(__expf(-fabsf(z))); }

__device__ __forceinline__ float dmll_color(float xi, float mi, float ls)
{
    float centered = xi - mi;
    float inv  = __expf(-ls);
    float plus = inv * (centered + 1.f / 255.f);
    float minv = inv * (centered - 1.f / 255.f);
    float sigp = 1.f / (1.f + __expf(-plus));
    float sigm = 1.f / (1.f + __expf(-minv));
    float cdf_delta = sigp - sigm;
    float log_cdf_plus = plus - sp_(plus);
    float log_om = -sp_(minv);
    float mid = inv * centered;
    float log_pdf_mid = mid - ls - 2.f * sp_(mid);
    float inner = (cdf_delta > 1e-5f) ? __logf(fmaxf(cdf_delta, 1e-12f))
                                      : (log_pdf_mid - 4.852030263919617f); // log(128)
    return (xi < -0.999f) ? log_cdf_plus : ((xi > 0.999f) ? log_om : inner);
}

__global__ void dmll_k(const float* __restrict__ xp, const float* __restrict__ l,
                       float* __restrict__ out)
{
    int pix = blockIdx.x * blockDim.x + threadIdx.x;
    if (pix >= 4 * 32 * 32) return;
    int b = pix >> 10;
    const float* lp = l + (size_t)pix * 1000;
    float x0 = xp[pix * 4 + 0], x1 = xp[pix * 4 + 1], x2 = xp[pix * 4 + 2];

    float mx = -1e30f;
    for (int m = 0; m < 100; ++m) mx = fmaxf(mx, lp[m]);
    float se = 0.f;
    for (int m = 0; m < 100; ++m) se += __expf(lp[m] - mx);
    float lse = mx + __logf(se);

    float lparr[100];
    float best = -1e30f;
    for (int m = 0; m < 100; ++m) {
        float mean0 = lp[100 + m];
        float mean1 = lp[100 + 300 + m];
        float mean2 = lp[100 + 600 + m];
        float ls0 = fmaxf(lp[100 + 100 + m], -7.f);
        float ls1 = fmaxf(lp[100 + 400 + m], -7.f);
        float ls2 = fmaxf(lp[100 + 700 + m], -7.f);
        float c0 = tanhf(lp[100 + 200 + m]);
        float c1 = tanhf(lp[100 + 500 + m]);
        float c2 = tanhf(lp[100 + 800 + m]);
        float m2v = mean1 + c0 * x0;
        float m3v = mean2 + c1 * x0 + c2 * x1;
        float t = dmll_color(x0, mean0, ls0) + dmll_color(x1, m2v, ls1)
                + dmll_color(x2, m3v, ls2) + lp[m] - lse;
        lparr[m] = t;
        best = fmaxf(best, t);
    }
    float s = 0.f;
    for (int m = 0; m < 100; ++m) s += __expf(lparr[m] - best);
    atomicAdd(&out[b], best + __logf(s));
}

// ---------------------------------------------------------------------------
// Host orchestration
// ---------------------------------------------------------------------------
struct ConvP { const float* b; const float* w; };
struct ResP  { const float *cond_b, *cond_w, *cin_b, *cin_w, *cout_b, *cout_w,
                           *nin_b, *nin_w; };

static inline int G(int n) { return (n + 255) / 256; }

static _Float16* g_whbuf = nullptr;  // set per-launch before use

static void launch_conv(hipStream_t st, const float* X, const float* Wt,
                        const float* bias, float* Y, int Bn, int H, int W,
                        int Cx, int act, int kh, int kw, int pt, int pl,
                        int stride, int Cout, int accum)
{
    int Cin = (act == 1) ? 2 * Cx : Cx;
    int K = kh * kw * Cin;
    int KC = K * Cout;
    w2h_k<<<G(KC), 256, 0, st>>>(Wt, g_whbuf, KC);
    int Ho = H / stride, Wo = W / stride;
    int M = Bn * Ho * Wo;
    dim3 grid((M + 127) / 128, (Cout + 31) / 32);
    conv_wmma_k<<<grid, 256, LDS_BYTES, st>>>(X, g_whbuf, bias, Y, Bn, H, W, Cx,
                                              Cin, kh, kw, pt, pl, stride,
                                              Ho, Wo, Cout, act, accum);
}

extern "C" void kernel_launch(void* const* d_in, const int* in_sizes, int n_in,
                              void* d_out, int out_size, void* d_ws, size_t ws_size,
                              hipStream_t stream)
{
    (void)in_sizes; (void)n_in; (void)ws_size;
    const float* x_in  = (const float*)d_in[0];   // (4,3,32,32)
    const float* hcond = (const float*)d_in[1];   // (4,64)

    // ---- walk param pytree leaves (sorted-key dict order, lists in order) --
    int pi = 2;
    auto nxt = [&]() -> const float* { return (const float*)d_in[pi++]; };
    auto getConv = [&](ConvP& c) { c.b = nxt(); c.w = nxt(); };
    auto getRes  = [&](ResP& r, bool skip) {
        r.cond_b = nxt(); r.cond_w = nxt();
        r.cin_b  = nxt(); r.cin_w  = nxt();
        r.cout_b = nxt(); r.cout_w = nxt();
        if (skip) { r.nin_b = nxt(); r.nin_w = nxt(); }
        else      { r.nin_b = nullptr; r.nin_w = nullptr; }
    };

    ResP dnU[3][6], dnUL[3][6];
    const int ndn[3] = {5, 6, 6};
    for (int i = 0; i < 3; ++i) {                 // 'dn'
        for (int j = 0; j < ndn[i]; ++j) getRes(dnU[i][j], true);
        for (int j = 0; j < ndn[i]; ++j) getRes(dnUL[i][j], true);
    }
    ConvP downU[2], downUL[2];
    for (int i = 0; i < 2; ++i) getConv(downU[i]);    // 'down_u'
    for (int i = 0; i < 2; ++i) getConv(downUL[i]);   // 'down_ul'
    ConvP ninOut;  getConv(ninOut);                   // 'nin_out'
    ConvP uInit;   getConv(uInit);                    // 'u_init'
    ConvP ulInitA; getConv(ulInitA);                  // 'ul_init_a'
    ConvP ulInitB; getConv(ulInitB);                  // 'ul_init_b'
    ResP upU[3][5], upUL[3][5];
    for (int i = 0; i < 3; ++i) {                 // 'up'
        for (int j = 0; j < 5; ++j) getRes(upU[i][j], false);
        for (int j = 0; j < 5; ++j) getRes(upUL[i][j], true);
    }
    ConvP upDecU[2], upDecUL[2];
    for (int i = 0; i < 2; ++i) getConv(upDecU[i]);   // 'up_u'
    for (int i = 0; i < 2; ++i) getConv(upDecUL[i]);  // 'up_ul'

    // ---- workspace bump allocator --------------------------------------
    char* base = (char*)d_ws;
    size_t off = 0;
    auto alloc = [&](size_t nfloats) -> float* {
        float* p = (float*)(base + off);
        off = (off + nfloats * sizeof(float) + 255) & ~(size_t)255;
        return p;
    };

    float* xp    = alloc(4 * 32 * 32 * 4);
    float* t1    = alloc(4096 * 160);
    float* t2    = alloc(4096 * 160);
    float* t3    = alloc(4096 * 160);
    float* tmpC  = alloc(4096 * 160);
    float* tmpC2 = alloc(4096 * 320);
    float* cat   = alloc(4096 * 320);
    float* condv = alloc(4 * 320);
    float* lbuf  = alloc((size_t)4096 * 1000);
    g_whbuf      = (_Float16*)alloc(320000);          // f16 weight staging (1.25MB)
    float* ub[2]  = { alloc(4096 * 160), alloc(4096 * 160) };
    float* ulb[2] = { alloc(4096 * 160), alloc(4096 * 160) };

    float* uS[20]; float* ulS[20]; int ut = 0, ult = 0;

    // ---- gated resnet helper -------------------------------------------
    auto gated = [&](const float* xv, const float* av, int aCx, const ResP& p,
                     int kh, int kw, int Hc, float* outv) {
        int M = 4 * Hc * Hc;
        launch_conv(stream, xv, p.cin_w, p.cin_b, tmpC, 4, Hc, Hc, 160,
                    /*act*/1, kh, kw, 1, 1, 1, 160, 0);
        if (av)
            launch_conv(stream, av, p.nin_w, p.nin_b, tmpC, 4, Hc, Hc, aCx,
                        /*act*/1, 1, 1, 0, 0, 1, 160, /*accum*/1);
        launch_conv(stream, tmpC, p.cout_w, p.cout_b, tmpC2, 4, Hc, Hc, 160,
                    /*act*/1, kh, kw, 1, 1, 1, 320, 0);
        cond_mm_k<<<G(4 * 320), 256, 0, stream>>>(hcond, p.cond_w, p.cond_b,
                                                  condv, 4, 64, 320);
        gate_k<<<G(M * 160), 256, 0, stream>>>(xv, tmpC2, condv, outv,
                                               M, 160, Hc * Hc);
    };

    // ---- input prep -----------------------------------------------------
    prep_k<<<G(4 * 32 * 32 * 4), 256, 0, stream>>>(x_in, xp, 4 * 32 * 32 * 4);

    // ---- init streams ---------------------------------------------------
    launch_conv(stream, xp, uInit.w, uInit.b, t1, 4, 32, 32, 4, 0, 2, 3, 1, 1, 1, 160, 0);
    float* u0 = alloc(4096 * 160);
    dshift_k<<<G(4096 * 160), 256, 0, stream>>>(t1, u0, 4, 32, 32, 160, 4096 * 160);
    launch_conv(stream, xp, ulInitA.w, ulInitA.b, t1, 4, 32, 32, 4, 0, 1, 3, 0, 1, 1, 160, 0);
    dshift_k<<<G(4096 * 160), 256, 0, stream>>>(t1, t2, 4, 32, 32, 160, 4096 * 160);
    launch_conv(stream, xp, ulInitB.w, ulInitB.b, t1, 4, 32, 32, 4, 0, 2, 1, 1, 0, 1, 160, 0);
    rshift_k<<<G(4096 * 160), 256, 0, stream>>>(t1, t3, 4, 32, 32, 160, 4096 * 160);
    float* ul0 = alloc(4096 * 160);
    add_k<<<G(4096 * 160), 256, 0, stream>>>(t2, t3, ul0, 4096 * 160);

    float* u = u0; float* ul = ul0;
    uS[ut++] = u; ulS[ult++] = ul;

    // ---- up pass --------------------------------------------------------
    for (int i = 0; i < 3; ++i) {
        int Hc = 32 >> i;
        int M = 4 * Hc * Hc;
        for (int j = 0; j < 5; ++j) {
            float* un = alloc(M * 160);
            gated(u, nullptr, 0, upU[i][j], 2, 3, Hc, un);
            u = un; uS[ut++] = u;
            float* uln = alloc(M * 160);
            gated(ul, u, 160, upUL[i][j], 2, 2, Hc, uln);
            ul = uln; ulS[ult++] = ul;
        }
        if (i < 2) {
            int M2 = M / 4;
            float* ud = alloc(M2 * 160);
            launch_conv(stream, u, downU[i].w, downU[i].b, ud, 4, Hc, Hc, 160,
                        0, 2, 3, 1, 1, 2, 160, 0);
            u = ud; uS[ut++] = u;
            float* uld = alloc(M2 * 160);
            launch_conv(stream, ul, downUL[i].w, downUL[i].b, uld, 4, Hc, Hc, 160,
                        0, 2, 2, 1, 1, 2, 160, 0);
            ul = uld; ulS[ult++] = ul;
        }
    }

    // ---- down pass ------------------------------------------------------
    u  = uS[--ut];
    ul = ulS[--ult];
    int upi = 0, ulpi = 0;
    auto unew  = [&]() -> float* { float* t = ub[upi];  upi ^= 1;  return t; };
    auto ulnew = [&]() -> float* { float* t = ulb[ulpi]; ulpi ^= 1; return t; };

    for (int i = 0; i < 3; ++i) {
        int Hc = (i == 0) ? 8 : ((i == 1) ? 16 : 32);
        int M = 4 * Hc * Hc;
        for (int j = 0; j < ndn[i]; ++j) {
            float* un = unew();
            gated(u, uS[--ut], 160, dnU[i][j], 2, 3, Hc, un);
            u = un;
            concat_k<<<G(M * 320), 256, 0, stream>>>(u, ulS[--ult], cat, M, 160);
            float* uln = ulnew();
            gated(ul, cat, 320, dnUL[i][j], 2, 2, Hc, uln);
            ul = uln;
        }
        if (i < 2) {
            int n = 4 * (2 * Hc) * (2 * Hc) * 160;
            float* un = unew();
            deconv_k<<<G(n), 256, 0, stream>>>(u, upDecU[i].w, upDecU[i].b, un,
                                               4, Hc, Hc, 160, 2, 3, n);
            u = un;
            float* uln = ulnew();
            deconv_k<<<G(n), 256, 0, stream>>>(ul, upDecUL[i].w, upDecUL[i].b, uln,
                                               4, Hc, Hc, 160, 2, 2, n);
            ul = uln;
        }
    }

    // ---- output head: nin(elu(ul)) -> l [4,32,32,1000] -------------------
    launch_conv(stream, ul, ninOut.w, ninOut.b, lbuf, 4, 32, 32, 160,
                /*act*/2, 1, 1, 0, 0, 1, 1000, 0);

    // ---- DMLL log-prob --------------------------------------------------
    zero_k<<<1, 256, 0, stream>>>((float*)d_out, out_size);
    dmll_k<<<G(4 * 32 * 32), 256, 0, stream>>>(xp, lbuf, (float*)d_out);
}
